// AxialAttention_14559939133608
// MI455X (gfx1250) — matrix operand order repaired and checked by reading, not compiled, not run on hardware
//
#include <hip/hip_runtime.h>
#include <hip/hip_bf16.h>

#define B_ 8
#define H_ 8
#define N_ 1024
#define C_ 512
#define D_ 64

typedef _Float16 half_t;
typedef __attribute__((ext_vector_type(16))) _Float16 v16h;
typedef __attribute__((ext_vector_type(8)))  _Float16 h8v;
typedef __attribute__((ext_vector_type(8)))  float    v8f;

// ---------------------------------------------------------------------------
// WMMA fragment helpers (V_WMMA_F32_16X16X32_F16, wave32)
// A (16xK tile, row-major, rows = M):  lane L: m = L%16, h = L/16
//   elements 0..7  = A[m, 8h .. 8h+7]   elements 8..15 = A[m, 16+8h .. 16+8h+7]
// B loaded from B^T (16xK tile, row-major, rows = N): lane L: n = L%16, h = L/16
//   elements 0..7  = B^T[n, 8h .. 8h+7]   elements 8..15 = B^T[n, 16+8h .. 16+8h+7]
// C/D: element r -> (m = r + 8*(L/16), n = L%16)
// ---------------------------------------------------------------------------
__device__ __forceinline__ v16h load_afrag(const half_t* p0, int ld) {
  const int lane = threadIdx.x & 31;
  const half_t* p = p0 + (lane & 15) * ld + ((lane >> 4) << 3);
  h8v lo = *reinterpret_cast<const h8v*>(p);
  h8v hi = *reinterpret_cast<const h8v*>(p + 16);
  v16h f;
#pragma unroll
  for (int e = 0; e < 8; ++e) { f[e] = lo[e]; f[e + 8] = hi[e]; }
  return f;
}

__device__ __forceinline__ v16h load_bfrag(const half_t* p0, int ld) {
  const int lane = threadIdx.x & 31;
  const half_t* p = p0 + (lane & 15) * ld + ((lane >> 4) << 3);
  h8v lo = *reinterpret_cast<const h8v*>(p);
  h8v hi = *reinterpret_cast<const h8v*>(p + 16);
  v16h f;
#pragma unroll
  for (int e = 0; e < 8; ++e) { f[e] = lo[e]; f[e + 8] = hi[e]; }
  return f;
}

__device__ __forceinline__ v8f wmma16(v16h a, v16h b, v8f c) {
  return __builtin_amdgcn_wmma_f32_16x16x32_f16(false, a, false, b, (short)0, c,
                                                false, false);
}

// ---------------------------------------------------------------------------
// K0: fp32 -> fp16 conversions + weight transposes (B^T layouts for WMMA)
// ---------------------------------------------------------------------------
__global__ void k_convert(const float* __restrict__ x,
                          const float* __restrict__ Wq, const float* __restrict__ Wk,
                          const float* __restrict__ Wv, const float* __restrict__ Wo,
                          const float* __restrict__ Erq, const float* __restrict__ Erk,
                          const float* __restrict__ Erv,
                          half_t* xh, half_t* WqT, half_t* WkT, half_t* WvT,
                          half_t* WoT, half_t* ErqH, half_t* ErkH, half_t* ErvT) {
  const int idx = blockIdx.x * blockDim.x + threadIdx.x;
  const int stride = gridDim.x * blockDim.x;
  const int NX = B_ * N_ * C_;
  for (int i = idx; i < NX; i += stride) xh[i] = (half_t)x[i];
  const int NW = C_ * C_;
  for (int i = idx; i < NW; i += stride) {
    int r = i >> 9, c = i & 511;         // WT[r][c] = W[c][r]
    WqT[i] = (half_t)Wq[c * C_ + r];
    WkT[i] = (half_t)Wk[c * C_ + r];
    WvT[i] = (half_t)Wv[c * C_ + r];
    WoT[i] = (half_t)Wo[c * C_ + r];
  }
  const int NE = N_ * D_;
  for (int i = idx; i < NE; i += stride) {
    ErqH[i] = (half_t)Erq[i];
    ErkH[i] = (half_t)Erk[i];
    int r = i >> 6, c = i & 63;
    ErvT[c * N_ + r] = (half_t)Erv[i];   // Er_v^T [64,1024]
  }
}

// ---------------------------------------------------------------------------
// K1: projection GEMM  [8192,512] x [512,512] -> q/k ([B,H,N,64]) or vT ([B,H,64,N])
// 32x32 register-blocked per wave: 4 WMMAs per 4 fragment loads.
// ---------------------------------------------------------------------------
__global__ void k_proj(const half_t* __restrict__ A, const half_t* __restrict__ Bt,
                       half_t* __restrict__ out, int mode) {
  const int wave = threadIdx.x >> 5;
  const int rt = blockIdx.x;                    // 0..255 (32-row tiles)
  const int ct = blockIdx.y * 4 + wave;         // 0..15  (32-col tiles)
  const half_t* a0p = A + rt * 32 * C_;
  const half_t* b0p = Bt + ct * 32 * C_;
  v8f accs[2][2] = {};
#pragma unroll 2
  for (int kk = 0; kk < C_; kk += 32) {
    v16h a0 = load_afrag(a0p + kk, C_);
    v16h a1 = load_afrag(a0p + 16 * C_ + kk, C_);
    v16h b0 = load_bfrag(b0p + kk, C_);
    v16h b1 = load_bfrag(b0p + 16 * C_ + kk, C_);
    accs[0][0] = wmma16(a0, b0, accs[0][0]);
    accs[0][1] = wmma16(a0, b1, accs[0][1]);
    accs[1][0] = wmma16(a1, b0, accs[1][0]);
    accs[1][1] = wmma16(a1, b1, accs[1][1]);
  }
  const int lane = threadIdx.x & 31;
  const int n = lane & 15, mo = (lane >> 4) * 8;
#pragma unroll
  for (int rh = 0; rh < 2; ++rh) {
#pragma unroll
    for (int ch = 0; ch < 2; ++ch) {
      const int colg = ct * 32 + ch * 16 + n;
      const int h = colg >> 6, d = colg & 63;
#pragma unroll
      for (int r = 0; r < 8; ++r) {
        int tok = rt * 32 + rh * 16 + mo + r;
        int b = tok >> 10, tk = tok & (N_ - 1);
        half_t val = (half_t)accs[rh][ch][r];
        if (mode == 0)
          out[(((b * H_ + h) * N_ + tk) * D_) + d] = val;      // [B,H,N,64]
        else
          out[(((b * H_ + h) * D_ + d) * N_) + tk] = val;      // [B,H,64,N]
      }
    }
  }
}

// ---------------------------------------------------------------------------
// K2 (per batch): S = scale * q k^T (fp16), bias = skew(q Er_q^T + k Er_k^T)
// scatter-written into its skewed destination (fp16). The j==i+1 diagonal of
// bias is never written and is masked to 0 in the softmax kernel.
// A-fragments (q,k rows) are hoisted and reused over 4 column tiles.
// ---------------------------------------------------------------------------
__global__ void k_logits(const half_t* __restrict__ q, const half_t* __restrict__ k,
                         const half_t* __restrict__ ErqH, const half_t* __restrict__ ErkH,
                         half_t* __restrict__ S, half_t* __restrict__ bias) {
  const int wave = threadIdx.x >> 5;
  const int jg = blockIdx.x * 4 + wave;   // col tile group 0..15 (4 tiles each)
  const int it = blockIdx.y;              // row tile 0..63
  const int h  = blockIdx.z;
  const half_t* qh = q + h * N_ * D_;
  const half_t* kh = k + h * N_ * D_;
  const v16h aq0 = load_afrag(qh + it * 16 * D_, D_);
  const v16h aq1 = load_afrag(qh + it * 16 * D_ + 32, D_);
  const v16h ak0 = load_afrag(kh + it * 16 * D_, D_);
  const v16h ak1 = load_afrag(kh + it * 16 * D_ + 32, D_);

  const int lane = threadIdx.x & 31;
  const int n = lane & 15, mo = (lane >> 4) * 8;
  half_t* Sh = S + (size_t)h * N_ * N_;
  half_t* Bh = bias + (size_t)h * N_ * N_;

#pragma unroll
  for (int t = 0; t < 4; ++t) {
    const int jt = jg * 4 + t;
    const half_t* bk = kh + jt * 16 * D_;
    const half_t* eq = ErqH + jt * 16 * D_;
    const half_t* ek = ErkH + jt * 16 * D_;
    v8f accS = {}, accB = {};
    accS = wmma16(aq0, load_bfrag(bk, D_), accS);
    accS = wmma16(aq1, load_bfrag(bk + 32, D_), accS);
    accB = wmma16(aq0, load_bfrag(eq, D_), accB);
    accB = wmma16(aq1, load_bfrag(eq + 32, D_), accB);
    accB = wmma16(ak0, load_bfrag(ek, D_), accB);
    accB = wmma16(ak1, load_bfrag(ek + 32, D_), accB);
#pragma unroll
    for (int r = 0; r < 8; ++r) {
      int i = it * 16 + mo + r;
      int j = jt * 16 + n;
      Sh[i * N_ + j] = (half_t)(accS[r] * 0.125f);   // scale = d^-0.5
      int tt = j - (N_ - 1 - i);                     // skew scatter (m = j)
      half_t bv = (half_t)accB[r];
      if (tt >= 0)      Bh[i * N_ + tt] = bv;
      else if (i >= 1)  Bh[(i - 1) * N_ + (j + i + 1)] = bv;
      // (i==0, tt<0) values are dropped by the skew
    }
  }
}

// ---------------------------------------------------------------------------
// K3 (per batch): row softmax of S + bias -> P (fp16), and emit skew(P)
// skew(P)[i,:] = [ P[i, N-1-i : N] , 0 , P[i+1, 0 : N-i-2] ]  (written inversely)
// ---------------------------------------------------------------------------
__global__ void k_softmax(const half_t* __restrict__ S, const half_t* __restrict__ bias,
                          half_t* __restrict__ P, half_t* __restrict__ skewP) {
  const int i = blockIdx.x, h = blockIdx.y, tid = threadIdx.x;
  const half_t* Srow = S    + ((size_t)h * N_ + i) * N_;
  const half_t* Brow = bias + ((size_t)h * N_ + i) * N_;
  half_t* Prow = P + ((size_t)h * N_ + i) * N_;
  half_t* SPh  = skewP + (size_t)h * N_ * N_;

  __shared__ float red1[8], red2[8];
  float v[4];
  float mx = -1e30f;
#pragma unroll
  for (int c = 0; c < 4; ++c) {
    int j = tid + c * 256;
    float b = (j == i + 1) ? 0.f : (float)Brow[j];   // unwritten diag == 0
    v[c] = (float)Srow[j] + b;
    mx = fmaxf(mx, v[c]);
  }
#pragma unroll
  for (int o = 16; o > 0; o >>= 1) mx = fmaxf(mx, __shfl_xor(mx, o, 32));
  if ((tid & 31) == 0) red1[tid >> 5] = mx;
  __syncthreads();
  float m = red1[0];
#pragma unroll
  for (int w = 1; w < 8; ++w) m = fmaxf(m, red1[w]);

  float s = 0.f;
#pragma unroll
  for (int c = 0; c < 4; ++c) { v[c] = __expf(v[c] - m); s += v[c]; }
#pragma unroll
  for (int o = 16; o > 0; o >>= 1) s += __shfl_xor(s, o, 32);
  if ((tid & 31) == 0) red2[tid >> 5] = s;
  __syncthreads();
  float tot = 0.f;
#pragma unroll
  for (int w = 0; w < 8; ++w) tot += red2[w];
  const float inv = 1.f / tot;

#pragma unroll
  for (int c = 0; c < 4; ++c) {
    int j = tid + c * 256;
    half_t p = (half_t)(v[c] * inv);
    Prow[j] = p;
    if (j >= N_ - 1 - i)      SPh[i * N_ + (j - (N_ - 1 - i))] = p;
    else if (i >= 1)          SPh[(i - 1) * N_ + (j + i + 1)] = p;
  }
  if (tid == 0 && i < N_ - 1) SPh[i * N_ + (i + 1)] = (half_t)0.f;
}

// ---------------------------------------------------------------------------
// K4 (per batch): attOut = P @ v + skew(P) @ Er_v   (both K=1024 WMMA chains)
// 32x32 register-blocked per wave. v / Er_v supplied transposed (B^T row-major).
// ---------------------------------------------------------------------------
__global__ void k_pv(const half_t* __restrict__ P, const half_t* __restrict__ skewP,
                     const half_t* __restrict__ vT, const half_t* __restrict__ ErvT,
                     half_t* __restrict__ attOut, int b) {
  const int wave = threadIdx.x >> 5;     // d half (32 cols) 0..1
  const int it = blockIdx.x;             // 32-row tile 0..31
  const int h  = blockIdx.y;
  const half_t* Ph  = P     + (size_t)h * N_ * N_ + it * 32 * N_;
  const half_t* SPh = skewP + (size_t)h * N_ * N_ + it * 32 * N_;
  const half_t* vTh = vT + (((size_t)b * H_ + h) * D_ + wave * 32) * N_;
  const half_t* EvT = ErvT + wave * 32 * N_;
  v8f accs[2][2] = {};
  for (int kk = 0; kk < N_; kk += 32) {
    v16h p0 = load_afrag(Ph + kk, N_);
    v16h p1 = load_afrag(Ph + 16 * N_ + kk, N_);
    v16h v0 = load_bfrag(vTh + kk, N_);
    v16h v1 = load_bfrag(vTh + 16 * N_ + kk, N_);
    accs[0][0] = wmma16(p0, v0, accs[0][0]);
    accs[0][1] = wmma16(p0, v1, accs[0][1]);
    accs[1][0] = wmma16(p1, v0, accs[1][0]);
    accs[1][1] = wmma16(p1, v1, accs[1][1]);
    v16h s0 = load_afrag(SPh + kk, N_);
    v16h s1 = load_afrag(SPh + 16 * N_ + kk, N_);
    v16h e0 = load_bfrag(EvT + kk, N_);
    v16h e1 = load_bfrag(EvT + 16 * N_ + kk, N_);
    accs[0][0] = wmma16(s0, e0, accs[0][0]);
    accs[0][1] = wmma16(s0, e1, accs[0][1]);
    accs[1][0] = wmma16(s1, e0, accs[1][0]);
    accs[1][1] = wmma16(s1, e1, accs[1][1]);
  }
  const int lane = threadIdx.x & 31;
  const int n = lane & 15, mo = (lane >> 4) * 8;
#pragma unroll
  for (int rh = 0; rh < 2; ++rh) {
#pragma unroll
    for (int ch = 0; ch < 2; ++ch) {
      const int col = h * D_ + wave * 32 + ch * 16 + n;
#pragma unroll
      for (int r = 0; r < 8; ++r) {
        int i = it * 32 + rh * 16 + mo + r;
        attOut[((size_t)(b * N_ + i)) * C_ + col] = (half_t)accs[rh][ch][r];
      }
    }
  }
}

// ---------------------------------------------------------------------------
// K5: out = attOut @ Wo + bo   (fp32 result), 32x32 register-blocked
// ---------------------------------------------------------------------------
__global__ void k_outproj(const half_t* __restrict__ A, const half_t* __restrict__ WoT,
                          const float* __restrict__ bo, float* __restrict__ out) {
  const int wave = threadIdx.x >> 5;
  const int rt = blockIdx.x;                    // 0..255
  const int ct = blockIdx.y * 4 + wave;         // 0..15
  const half_t* a0p = A + rt * 32 * C_;
  const half_t* b0p = WoT + ct * 32 * C_;
  v8f accs[2][2] = {};
#pragma unroll 2
  for (int kk = 0; kk < C_; kk += 32) {
    v16h a0 = load_afrag(a0p + kk, C_);
    v16h a1 = load_afrag(a0p + 16 * C_ + kk, C_);
    v16h b0 = load_bfrag(b0p + kk, C_);
    v16h b1 = load_bfrag(b0p + 16 * C_ + kk, C_);
    accs[0][0] = wmma16(a0, b0, accs[0][0]);
    accs[0][1] = wmma16(a0, b1, accs[0][1]);
    accs[1][0] = wmma16(a1, b0, accs[1][0]);
    accs[1][1] = wmma16(a1, b1, accs[1][1]);
  }
  const int lane = threadIdx.x & 31;
  const int n = lane & 15, mo = (lane >> 4) * 8;
#pragma unroll
  for (int rh = 0; rh < 2; ++rh) {
#pragma unroll
    for (int ch = 0; ch < 2; ++ch) {
      const int col = ct * 32 + ch * 16 + n;
      const float bias = bo[col];
#pragma unroll
      for (int r = 0; r < 8; ++r) {
        int row = rt * 32 + rh * 16 + mo + r;
        out[(size_t)row * C_ + col] = accs[rh][ch][r] + bias;
      }
    }
  }
}

// ---------------------------------------------------------------------------
extern "C" void kernel_launch(void* const* d_in, const int* in_sizes, int n_in,
                              void* d_out, int out_size, void* d_ws, size_t ws_size,
                              hipStream_t stream) {
  (void)in_sizes; (void)n_in; (void)out_size; (void)ws_size;
  const float* x   = (const float*)d_in[0];
  const float* Wq  = (const float*)d_in[1];
  const float* Wk  = (const float*)d_in[2];
  const float* Wv  = (const float*)d_in[3];
  const float* Erq = (const float*)d_in[4];
  const float* Erk = (const float*)d_in[5];
  const float* Erv = (const float*)d_in[6];
  const float* Wo  = (const float*)d_in[7];
  const float* bo  = (const float*)d_in[8];

  char* w = (char*)d_ws;
  size_t off = 0;
  auto take = [&](size_t bytes) -> char* {
    char* p = w + off;
    off += (bytes + 255) & ~(size_t)255;
    return p;
  };
  half_t* xh   = (half_t*)take((size_t)B_ * N_ * C_ * 2);
  half_t* WqT  = (half_t*)take((size_t)C_ * C_ * 2);
  half_t* WkT  = (half_t*)take((size_t)C_ * C_ * 2);
  half_t* WvT  = (half_t*)take((size_t)C_ * C_ * 2);
  half_t* WoT  = (half_t*)take((size_t)C_ * C_ * 2);
  half_t* ErqH = (half_t*)take((size_t)N_ * D_ * 2);
  half_t* ErkH = (half_t*)take((size_t)N_ * D_ * 2);
  half_t* ErvT = (half_t*)take((size_t)D_ * N_ * 2);
  half_t* qbuf = (half_t*)take((size_t)B_ * H_ * N_ * D_ * 2);
  half_t* kbuf = (half_t*)take((size_t)B_ * H_ * N_ * D_ * 2);
  half_t* vT   = (half_t*)take((size_t)B_ * H_ * D_ * N_ * 2);
  half_t* attO = (half_t*)take((size_t)B_ * N_ * C_ * 2);
  half_t* Sbuf = (half_t*)take((size_t)H_ * N_ * N_ * 2);   // per-batch
  half_t* bias = (half_t*)take((size_t)H_ * N_ * N_ * 2);   // per-batch
  half_t* Pbuf = (half_t*)take((size_t)H_ * N_ * N_ * 2);   // per-batch
  half_t* sPbf = (half_t*)take((size_t)H_ * N_ * N_ * 2);   // per-batch

  k_convert<<<dim3(2048), 256, 0, stream>>>(x, Wq, Wk, Wv, Wo, Erq, Erk, Erv,
                                            xh, WqT, WkT, WvT, WoT, ErqH, ErkH, ErvT);

  k_proj<<<dim3(256, 4), 128, 0, stream>>>(xh, WqT, qbuf, 0);
  k_proj<<<dim3(256, 4), 128, 0, stream>>>(xh, WkT, kbuf, 0);
  k_proj<<<dim3(256, 4), 128, 0, stream>>>(xh, WvT, vT, 1);

  for (int b = 0; b < B_; ++b) {
    const half_t* qb = qbuf + (size_t)b * H_ * N_ * D_;
    const half_t* kb = kbuf + (size_t)b * H_ * N_ * D_;
    k_logits<<<dim3(4, 64, 8), 128, 0, stream>>>(qb, kb, ErqH, ErkH, Sbuf, bias);
    k_softmax<<<dim3(1024, 8), 256, 0, stream>>>(Sbuf, bias, Pbuf, sPbf);
    k_pv<<<dim3(32, 8), 64, 0, stream>>>(Pbuf, sPbf, vT, ErvT, attO, b);
  }

  k_outproj<<<dim3(256, 4), 128, 0, stream>>>(attO, WoT, bo, (float*)d_out);
}
